// AsynchronousKANLayer_4922032521493
// MI455X (gfx1250) — matrix operand order repaired and checked by reading, not compiled, not run on hardware
//
#include <hip/hip_runtime.h>

typedef unsigned int u32;
typedef unsigned short u16;
typedef __bf16 bf16_t;
typedef __attribute__((ext_vector_type(16))) bf16_t v16bf;
typedef __attribute__((ext_vector_type(8)))  float  v8f;
typedef int i32x4 __attribute__((vector_size(16)));

union Frag { uint4 q[2]; v16bf v; };

// ---- CDNA5 async global->LDS path (guarded; falls back to sync loads) ----
#if defined(__has_builtin)
#if __has_builtin(__builtin_amdgcn_global_load_async_to_lds_b128)
#define HAVE_ASYNC_LDS 1
#endif
#if __has_builtin(__builtin_amdgcn_s_wait_asynccnt)
#define HAVE_WAIT_ASYNC_BUILTIN 1
#endif
#endif
#ifndef HAVE_ASYNC_LDS
#define HAVE_ASYNC_LDS 0
#endif

// builtin signature (from hipcc diagnostic): (int4 AS1*, int4 AS3*, Ii, Ii)
#define GAS1(p) ((__attribute__((address_space(1))) i32x4*)(void*)(p))
#define LAS3(p) ((__attribute__((address_space(3))) i32x4*)(void*)(p))

__device__ __forceinline__ void async_wait0() {
#if defined(HAVE_WAIT_ASYNC_BUILTIN)
    __builtin_amdgcn_s_wait_asynccnt(0);
#else
    asm volatile("s_wait_asynccnt 0" ::: "memory");
#endif
}

__device__ __forceinline__ u16 f2bf(float f) {
    u32 u = __builtin_bit_cast(u32, f);
    u32 r = u + 0x7FFFu + ((u >> 16) & 1u);   // round-to-nearest-even
    return (u16)(r >> 16);
}

// butterfly max within each 16-lane half (ds_swizzle, no index VGPR)
__device__ __forceinline__ float rowmax16(float x) {
    float y;
    y = __builtin_bit_cast(float, __builtin_amdgcn_ds_swizzle(__builtin_bit_cast(int, x), 0x041F));
    x = fmaxf(x, y);
    y = __builtin_bit_cast(float, __builtin_amdgcn_ds_swizzle(__builtin_bit_cast(int, x), 0x081F));
    x = fmaxf(x, y);
    y = __builtin_bit_cast(float, __builtin_amdgcn_ds_swizzle(__builtin_bit_cast(int, x), 0x101F));
    x = fmaxf(x, y);
    y = __builtin_bit_cast(float, __builtin_amdgcn_ds_swizzle(__builtin_bit_cast(int, x), 0x201F));
    x = fmaxf(x, y);
    return x;
}

// ---------------------------------------------------------------------------
// fp32 -> bf16 conversion (weights)
// ---------------------------------------------------------------------------
__global__ void cvt_f32_bf16_kernel(const float* __restrict__ in,
                                    u16* __restrict__ out, int n) {
    int i = blockIdx.x * 256 + threadIdx.x;
    if (i < n) out[i] = f2bf(in[i]);
}

// ---------------------------------------------------------------------------
// LayerNorm over D=1024, output bf16. One block (256 thr) per row.
// ---------------------------------------------------------------------------
__global__ void __launch_bounds__(256)
ln_bf16_kernel(const float* __restrict__ x, const float* __restrict__ g,
               const float* __restrict__ beta, u16* __restrict__ out) {
    const int D = 1024;
    const int row = blockIdx.x;
    const int t = threadIdx.x;
    __shared__ float red[256];

    float4 xv = ((const float4*)(x + (size_t)row * D))[t];
    float s = xv.x + xv.y + xv.z + xv.w;
    red[t] = s;
    __syncthreads();
    for (int o = 128; o > 0; o >>= 1) { if (t < o) red[t] += red[t + o]; __syncthreads(); }
    float mu = red[0] * (1.0f / D);
    __syncthreads();

    float d0 = xv.x - mu, d1 = xv.y - mu, d2 = xv.z - mu, d3 = xv.w - mu;
    red[t] = d0 * d0 + d1 * d1 + d2 * d2 + d3 * d3;
    __syncthreads();
    for (int o = 128; o > 0; o >>= 1) { if (t < o) red[t] += red[t + o]; __syncthreads(); }
    float rs = rsqrtf(red[0] * (1.0f / D) + 1e-5f);

    float4 gv = ((const float4*)g)[t];
    float4 bv = ((const float4*)beta)[t];
    u16* o = out + (size_t)row * D + t * 4;
    o[0] = f2bf(d0 * rs * gv.x + bv.x);
    o[1] = f2bf(d1 * rs * gv.y + bv.y);
    o[2] = f2bf(d2 * rs * gv.z + bv.z);
    o[3] = f2bf(d3 * rs * gv.w + bv.w);
}

// ---------------------------------------------------------------------------
// WMMA bf16 GEMM:  C[M,N] = A[M,K] @ W[N,K]^T + bias  (+ epilogue)
// Block tile 128x128, BK=32, 8 waves each computing 32(M)x64(N).
// ---------------------------------------------------------------------------
#define EP_BF16      1   // out bf16 (QKV projection)
#define EP_RES_F32   2   // out f32 = C + bias + residual (O-proj)
#define EP_GELU_BF16 3   // out bf16 = gelu(C + bias)     (MLP up)
#define EP_FINAL     4   // out f32 = ternary(spike(C+bias+res)) (MLP down)

template <int EP>
__global__ void __launch_bounds__(256)
gemm_wmma_bf16(const u16* __restrict__ A, const u16* __restrict__ W,
               const float* __restrict__ bias, const float* __restrict__ res,
               float* __restrict__ outf, u16* __restrict__ outbf,
               int M, int N, int K) {
    constexpr int BM = 128, BN = 128, BK = 32;
    constexpr int ASld = 48;    // As row stride (bf16 elems, 96B: 16B-aligned)
    constexpr int BSld = 136;   // Bt row stride (272B: 16B-aligned)
    __shared__ u16 As[BM * ASld];
    __shared__ u16 Bt[BK * BSld];   // stored transposed: [k][n]

    const int tid   = threadIdx.x;
    const int lane  = tid & 31;
    const int wid   = tid >> 5;
    const int wm    = wid & 3;       // 4 waves along M
    const int wn    = wid >> 2;      // 2 waves along N
    const int m_blk = blockIdx.y * BM;
    const int n_blk = blockIdx.x * BN;

    const int lrow  = lane & 15;
    const int lhalf = lane >> 4;
    const int khalf = lhalf * 8;

    v8f zero = {0, 0, 0, 0, 0, 0, 0, 0};
    v8f acc[2][4];
#pragma unroll
    for (int i = 0; i < 2; i++)
#pragma unroll
        for (int j = 0; j < 4; j++) acc[i][j] = zero;

    const int arow = tid >> 1;            // 0..127
    const int akc  = (tid & 1) * 16;      // 0 or 16

    for (int k0 = 0; k0 < K; k0 += BK) {
        {   // stage A tile [128 x 32] row-major: async DMA to LDS
            const u16* src = A + (size_t)(m_blk + arow) * K + k0 + akc;
            u16* dst = &As[arow * ASld + akc];
#if HAVE_ASYNC_LDS
            __builtin_amdgcn_global_load_async_to_lds_b128(GAS1(src), LAS3(dst), 0, 0);
            __builtin_amdgcn_global_load_async_to_lds_b128(GAS1(src + 8), LAS3(dst + 8), 0, 0);
#else
            uint4 v0 = ((const uint4*)src)[0];
            uint4 v1 = ((const uint4*)src)[1];
            *(uint4*)dst       = v0;
            *(uint4*)(dst + 8) = v1;
#endif
        }
        {   // stage W tile transposed -> Bt[k][n] (layout change: manual)
            const u16* src = W + (size_t)(n_blk + arow) * K + k0 + akc;
            union { uint4 q[2]; u16 s[16]; } tmp;
            tmp.q[0] = ((const uint4*)src)[0];
            tmp.q[1] = ((const uint4*)src)[1];
#pragma unroll
            for (int i = 0; i < 16; i++) Bt[(akc + i) * BSld + arow] = tmp.s[i];
        }
#if HAVE_ASYNC_LDS
        async_wait0();
#endif
        __syncthreads();

        Frag a[2], b[4];
#pragma unroll
        for (int mi = 0; mi < 2; mi++) {
            const u16* p = &As[(wm * 32 + mi * 16 + lrow) * ASld + khalf];
            a[mi].q[0] = *(const uint4*)p;
            a[mi].q[1] = *(const uint4*)(p + 16);
        }
#pragma unroll
        for (int ni = 0; ni < 4; ni++) {
            const u16* p = &Bt[wn * 64 + ni * 16 + lrow];
#pragma unroll
            for (int i = 0; i < 16; i++)
                b[ni].v[i] = __builtin_bit_cast(bf16_t, p[(khalf + (i & 7) + (i >> 3) * 16) * BSld]);
        }
#pragma unroll
        for (int mi = 0; mi < 2; mi++)
#pragma unroll
            for (int ni = 0; ni < 4; ni++)
                acc[mi][ni] = __builtin_amdgcn_wmma_f32_16x16x32_bf16(
                    false, a[mi].v, false, b[ni].v, (short)0, acc[mi][ni],
                    false, false);
        __syncthreads();
    }

#pragma unroll
    for (int mi = 0; mi < 2; mi++) {
#pragma unroll
        for (int ni = 0; ni < 4; ni++) {
            int n = n_blk + wn * 64 + ni * 16 + lrow;
            float bv = bias[n];
#pragma unroll
            for (int v = 0; v < 8; v++) {
                int m = m_blk + wm * 32 + mi * 16 + v + lhalf * 8;
                float val = acc[mi][ni][v] + bv;
                size_t idx = (size_t)m * N + n;
                if constexpr (EP == EP_BF16) {
                    outbf[idx] = f2bf(val);
                } else if constexpr (EP == EP_RES_F32) {
                    outf[idx] = val + res[idx];
                } else if constexpr (EP == EP_GELU_BF16) {
                    float gel = 0.5f * val * (1.0f + erff(val * 0.70710678f));
                    outbf[idx] = f2bf(gel);
                } else {  // EP_FINAL: spike + ternary
                    float y = val + res[idx];
                    float spike = (y > 1.0f) ? 1.0f : 0.0f;
                    float tern = (2.0f * spike - 1.0f) *
                                 ((fabsf(y) > 0.55f) ? 1.0f : 0.0f);
                    outf[idx] = tern;
                }
            }
        }
    }
}

// ---------------------------------------------------------------------------
// Flash attention. Grid (S/64, B*H). Block = 128 thr (4 waves x 16 rows).
// qkv: bf16 [S*B, 3*1024] (q | k | v per head h at col h*64).
// ctxout: bf16 [S*B, 1024].
// Row-sums of P computed on the matrix pipe: l += P @ ones.
// ---------------------------------------------------------------------------
__global__ void __launch_bounds__(128)
flash_attn_kernel(const u16* __restrict__ qkv, u16* __restrict__ ctxout) {
    constexpr int S = 2048, B = 4, T = 32;
    __shared__ u16 Kt[64 * 48];       // [dh][key] transposed
    __shared__ u16 Vs[32 * 72];       // [key][dh] row-major
    __shared__ u16 Pb[4 * 16 * 32];   // per-wave P tile (C->A relayout)

    const int tid   = threadIdx.x;
    const int lane  = tid & 31;
    const int w     = tid >> 5;
    const int lrow  = lane & 15;
    const int lhalf = lane >> 4;
    const int khalf = lhalf * 8;
    const int sblk  = blockIdx.x;
    const int b     = blockIdx.y >> 4;
    const int h     = blockIdx.y & 15;

    // Q fragments for this wave's 16 rows (Dh=64 -> two K-tiles of 32)
    Frag qf[2];
    {
        int s = sblk * 64 + w * 16 + lrow;
        const u16* qrow = qkv + (size_t)(s * B + b) * 3072 + h * 64;
#pragma unroll
        for (int kt = 0; kt < 2; kt++) {
            const u16* p = qrow + kt * 32 + khalf;
            qf[kt].q[0] = *(const uint4*)p;
            qf[kt].q[1] = *(const uint4*)(p + 16);
        }
    }

    // constant all-ones bf16 B-operand (for P @ ones row sums)
    Frag onesf;
#pragma unroll
    for (int i = 0; i < 2; i++) {
        onesf.q[i].x = 0x3f803f80u; onesf.q[i].y = 0x3f803f80u;
        onesf.q[i].z = 0x3f803f80u; onesf.q[i].w = 0x3f803f80u;
    }

    v8f zero = {0, 0, 0, 0, 0, 0, 0, 0};
    v8f ctx[4];
#pragma unroll
    for (int dt = 0; dt < 4; dt++) ctx[dt] = zero;
    v8f l_acc = zero;
    float mi[8];
#pragma unroll
    for (int v = 0; v < 8; v++) mi[v] = -1e30f;

    const int j   = tid >> 2;          // key within block: 0..31
    const int dhc = (tid & 3) * 16;    // dh chunk

    for (int t0 = 0; t0 < S; t0 += T) {
        __syncthreads();
        {   // stage K (transposed, manual) and V (async DMA) blocks
            int srow = (t0 + j) * B + b;
            const u16* ksrc = qkv + (size_t)srow * 3072 + 1024 + h * 64 + dhc;
            union { uint4 q[2]; u16 s[16]; } tmp;
            tmp.q[0] = ((const uint4*)ksrc)[0];
            tmp.q[1] = ((const uint4*)ksrc)[1];
#pragma unroll
            for (int i = 0; i < 16; i++) Kt[(dhc + i) * 48 + j] = tmp.s[i];
            const u16* vsrc = qkv + (size_t)srow * 3072 + 2048 + h * 64 + dhc;
            u16* vdst = &Vs[j * 72 + dhc];
#if HAVE_ASYNC_LDS
            __builtin_amdgcn_global_load_async_to_lds_b128(GAS1(vsrc), LAS3(vdst), 0, 0);
            __builtin_amdgcn_global_load_async_to_lds_b128(GAS1(vsrc + 8), LAS3(vdst + 8), 0, 0);
#else
            uint4 v0 = ((const uint4*)vsrc)[0];
            uint4 v1 = ((const uint4*)vsrc)[1];
            *(uint4*)vdst       = v0;
            *(uint4*)(vdst + 8) = v1;
#endif
        }
#if HAVE_ASYNC_LDS
        async_wait0();
#endif
        __syncthreads();

        // scores: 16 rows x 32 keys = two WMMA N-tiles, K-dim = Dh = 64
        v8f s0 = zero, s1 = zero;
#pragma unroll
        for (int kt = 0; kt < 2; kt++) {
            Frag b0, b1;
            const u16* p = &Kt[(kt * 32) * 48 + lrow];
#pragma unroll
            for (int i = 0; i < 16; i++) {
                int kk = khalf + (i & 7) + (i >> 3) * 16;
                b0.v[i] = __builtin_bit_cast(bf16_t, p[kk * 48]);
                b1.v[i] = __builtin_bit_cast(bf16_t, p[kk * 48 + 16]);
            }
            s0 = __builtin_amdgcn_wmma_f32_16x16x32_bf16(false, qf[kt].v, false,
                    b0.v, (short)0, s0, false, false);
            s1 = __builtin_amdgcn_wmma_f32_16x16x32_bf16(false, qf[kt].v, false,
                    b1.v, (short)0, s1, false, false);
        }
        s0 = s0 * 0.125f;   // 1/sqrt(64)
        s1 = s1 * 0.125f;

        // online softmax: rows in (vgpr v, lane-half), cols in 16 lanes
#pragma unroll
        for (int v = 0; v < 8; v++) {
            float rmax = rowmax16(fmaxf(s0[v], s1[v]));
            float newm = fmaxf(mi[v], rmax);
            float corr = __expf(mi[v] - newm);
            mi[v] = newm;
            float p0 = __expf(s0[v] - newm);
            float p1 = __expf(s1[v] - newm);
            l_acc[v] = l_acc[v] * corr;
#pragma unroll
            for (int dt = 0; dt < 4; dt++) ctx[dt][v] = ctx[dt][v] * corr;
            int rloc = v + lhalf * 8;
            Pb[w * 512 + rloc * 32 + lrow]      = f2bf(p0);
            Pb[w * 512 + rloc * 32 + 16 + lrow] = f2bf(p1);
        }
        asm volatile("s_wait_dscnt 0" ::: "memory");  // wave-local LDS RAW

        // reload P in A-fragment layout; ctx += P @ V ; l += P @ ones
        Frag pf;
        {
            const u16* p = &Pb[w * 512 + lrow * 32 + khalf];
            pf.q[0] = *(const uint4*)p;
            pf.q[1] = *(const uint4*)(p + 16);
        }
        l_acc = __builtin_amdgcn_wmma_f32_16x16x32_bf16(false, pf.v, false,
                    onesf.v, (short)0, l_acc, false, false);
#pragma unroll
        for (int dt = 0; dt < 4; dt++) {
            Frag bv;
            const u16* p = &Vs[dt * 16 + lrow];
#pragma unroll
            for (int i = 0; i < 16; i++)
                bv.v[i] = __builtin_bit_cast(bf16_t, p[(khalf + (i & 7) + (i >> 3) * 16) * 72]);
            ctx[dt] = __builtin_amdgcn_wmma_f32_16x16x32_bf16(false, pf.v, false,
                        bv.v, (short)0, ctx[dt], false, false);
        }
    }

    // normalize + write context (bf16) in [S*B, 1024] layout
#pragma unroll
    for (int v = 0; v < 8; v++) {
        int s = sblk * 64 + w * 16 + v + lhalf * 8;
        float inv = 1.0f / l_acc[v];
#pragma unroll
        for (int dt = 0; dt < 4; dt++) {
            float val = ctx[dt][v] * inv;
            ctxout[(size_t)(s * B + b) * 1024 + h * 64 + dt * 16 + lrow] =
                f2bf(val);
        }
    }
}

// ---------------------------------------------------------------------------
extern "C" void kernel_launch(void* const* d_in, const int* in_sizes, int n_in,
                              void* d_out, int out_size, void* d_ws,
                              size_t ws_size, hipStream_t stream) {
    const float* x     = (const float*)d_in[0];
    const float* ln1_g = (const float*)d_in[1];
    const float* ln1_b = (const float*)d_in[2];
    const float* ln2_g = (const float*)d_in[3];
    const float* ln2_b = (const float*)d_in[4];
    const float* w_qkv = (const float*)d_in[5];
    const float* b_qkv = (const float*)d_in[6];
    const float* w_o   = (const float*)d_in[7];
    const float* b_o   = (const float*)d_in[8];
    const float* w1    = (const float*)d_in[9];
    const float* b1    = (const float*)d_in[10];
    const float* w2    = (const float*)d_in[11];
    const float* b2    = (const float*)d_in[12];
    (void)in_sizes; (void)n_in; (void)out_size; (void)ws_size;

    const int S = 2048, B = 4, D = 1024, Hd = 4096, M = S * B;

    char* ws = (char*)d_ws;
    size_t off = 0;
    auto alloc = [&](size_t bytes) -> void* {
        void* p = ws + off;
        off += (bytes + 255) & ~(size_t)255;
        return p;
    };
    u16*   h_bf    = (u16*)alloc((size_t)M * D * 2);       // LN1 out, reused for LN2
    u16*   qkv_bf  = (u16*)alloc((size_t)M * 3 * D * 2);   // 48 MB
    u16*   ctx_bf  = (u16*)alloc((size_t)M * D * 2);       // 16 MB (contig after qkv)
    float* x1      = (float*)alloc((size_t)M * D * 4);
    u16*   wqkv_bf = (u16*)alloc((size_t)3 * D * D * 2);
    u16*   wo_bf   = (u16*)alloc((size_t)D * D * 2);
    u16*   w1_bf   = (u16*)alloc((size_t)Hd * D * 2);
    u16*   w2_bf   = (u16*)alloc((size_t)D * Hd * 2);
    u16*   mlp1_bf = qkv_bf;  // overlay dead qkv+ctx (50.3+16.7 MB = 64 MB)

    // 1. weights -> bf16
    cvt_f32_bf16_kernel<<<(3 * D * D + 255) / 256, 256, 0, stream>>>(w_qkv, wqkv_bf, 3 * D * D);
    cvt_f32_bf16_kernel<<<(D * D + 255) / 256, 256, 0, stream>>>(w_o, wo_bf, D * D);
    cvt_f32_bf16_kernel<<<(Hd * D + 255) / 256, 256, 0, stream>>>(w1, w1_bf, Hd * D);
    cvt_f32_bf16_kernel<<<(D * Hd + 255) / 256, 256, 0, stream>>>(w2, w2_bf, D * Hd);

    // 2. LN1
    ln_bf16_kernel<<<M, 256, 0, stream>>>(x, ln1_g, ln1_b, h_bf);

    // 3. QKV projection: [M,3072] bf16
    gemm_wmma_bf16<EP_BF16><<<dim3(3 * D / 128, M / 128), 256, 0, stream>>>(
        h_bf, wqkv_bf, b_qkv, nullptr, nullptr, qkv_bf, M, 3 * D, D);

    // 4. attention -> ctx bf16
    flash_attn_kernel<<<dim3(S / 64, B * 16), 128, 0, stream>>>(qkv_bf, ctx_bf);

    // 5. O-projection + residual -> x1 (f32)
    gemm_wmma_bf16<EP_RES_F32><<<dim3(D / 128, M / 128), 256, 0, stream>>>(
        ctx_bf, wo_bf, b_o, x, x1, nullptr, M, D, D);

    // 6. LN2
    ln_bf16_kernel<<<M, 256, 0, stream>>>(x1, ln2_g, ln2_b, h_bf);

    // 7. MLP up + exact GELU -> bf16
    gemm_wmma_bf16<EP_GELU_BF16><<<dim3(Hd / 128, M / 128), 256, 0, stream>>>(
        h_bf, w1_bf, b1, nullptr, nullptr, mlp1_bf, M, Hd, D);

    // 8. MLP down + residual + spike/ternary -> d_out (f32)
    gemm_wmma_bf16<EP_FINAL><<<dim3(D / 128, M / 128), 256, 0, stream>>>(
        mlp1_bf, w2_bf, b2, x1, (float*)d_out, nullptr, M, D, Hd);
}
